// CausalAttention2d_14413910245708
// MI455X (gfx1250) — matrix operand order repaired and checked by reading, not compiled, not run on hardware
//
#include <hip/hip_runtime.h>
#include <hip/hip_bf16.h>
#include <stdint.h>

typedef __bf16 bf16_t;
typedef __attribute__((ext_vector_type(16))) __bf16 v16bf;
typedef __attribute__((ext_vector_type(8)))  float  v8f;
typedef __attribute__((ext_vector_type(4)))  unsigned int v4u;
typedef __attribute__((ext_vector_type(8)))  int  v8i;
typedef __attribute__((ext_vector_type(4)))  int  v4i;

#define NH   8
#define HD   64
#define EDIM 512
#define CDIM 512
#define NTOK 4096
#define L2E  1.44269504088896340736f

union BFrag { v16bf v; uint4 q[2]; };

// Load a 16-element bf16 fragment from two 16-byte chunks (global or LDS).
static __device__ inline v16bf ld_frag2(const bf16_t* p0, const bf16_t* p1) {
    BFrag f;
    f.q[0] = *reinterpret_cast<const uint4*>(p0);
    f.q[1] = *reinterpret_cast<const uint4*>(p1);
    return f.v;
}

// Convert f32 runs p[0..7] and p[16..23] -> bf16 fragment (table K order).
static __device__ inline v16bf cvt16(const float* p) {
    float4 f0 = reinterpret_cast<const float4*>(p)[0];
    float4 f1 = reinterpret_cast<const float4*>(p)[1];
    float4 f2 = reinterpret_cast<const float4*>(p)[4];
    float4 f3 = reinterpret_cast<const float4*>(p)[5];
    v16bf r;
    r[0]=(bf16_t)f0.x; r[1]=(bf16_t)f0.y; r[2]=(bf16_t)f0.z; r[3]=(bf16_t)f0.w;
    r[4]=(bf16_t)f1.x; r[5]=(bf16_t)f1.y; r[6]=(bf16_t)f1.z; r[7]=(bf16_t)f1.w;
    r[8]=(bf16_t)f2.x; r[9]=(bf16_t)f2.y; r[10]=(bf16_t)f2.z; r[11]=(bf16_t)f2.w;
    r[12]=(bf16_t)f3.x; r[13]=(bf16_t)f3.y; r[14]=(bf16_t)f3.z; r[15]=(bf16_t)f3.w;
    return r;
}

static __device__ inline v8f wmma_bf16(v16bf a, v16bf b, v8f c) {
    return __builtin_amdgcn_wmma_f32_16x16x32_bf16(false, a, false, b, (short)0, c,
                                                   false, false);
}

// ---------------------------------------------------------------------------
// Tensor Data Mover helpers (D# per CDNA5 ISA ch.8 / CDNA5 ISA)
// ---------------------------------------------------------------------------
static __device__ inline v4u tdm_g0(uint32_t lds_addr, uint64_t gaddr) {
    v4u g;
    g[0] = 1u;                                   // count=1, user descriptor
    g[1] = lds_addr;                             // LDS byte address
    g[2] = (uint32_t)gaddr;                      // global_addr[31:0]
    g[3] = (uint32_t)((gaddr >> 32) & 0x01FFFFFFu) | (2u << 30);  // addr hi | type=2
    return g;
}

// data_size=2B; 2D tile (tile1=0 -> 1D). Dims/strides in elements.
static __device__ inline v8i tdm_g1(uint32_t td0, uint32_t td1,
                                    uint32_t tile0, uint32_t tile1,
                                    uint32_t stride0) {
    v8i g = (v8i)0;
    g[0] = (int)(1u << 16);                                          // data_size=2B
    g[1] = (int)((td0 & 0xFFFFu) << 16);                             // tensor_dim0 lo
    g[2] = (int)(((td0 >> 16) & 0xFFFFu) | ((td1 & 0xFFFFu) << 16)); // td0 hi|td1 lo
    g[3] = (int)(((td1 >> 16) & 0xFFFFu) | (tile0 << 16));           // td1 hi|tile0
    g[4] = (int)(tile1 & 0xFFFFu);                                   // tile1|tile2=0
    g[5] = (int)stride0;                                             // stride0[31:0]
    g[6] = 0;
    g[7] = 0;
    return g;
}

static __device__ inline void tdm_load(v4u g0, v8i g1) {
    v4i z4 = (v4i)0;
#if defined(__clang_major__) && (__clang_major__ >= 23)
    v8i z8 = (v8i)0;
    __builtin_amdgcn_tensor_load_to_lds(g0, g1, z4, z4, z8, 0);
#else
    __builtin_amdgcn_tensor_load_to_lds(g0, g1, z4, z4, 0);
#endif
}

static __device__ inline uint32_t lds_off(const void* p) {
    return (uint32_t)(size_t)p;   // LDS aperture: addr[31:0] is the LDS byte offset
}

// ---------------------------------------------------------------------------
// Projection: out[n,e] = sum_c in[b,c,n] * W[e,c] + bias[e]   (bf16 out)
// transposed==0: out stored (b,h,n,d); ==1: stored (b,h,d,n).
// Block = 256 threads (8 waves) computes a 64(n) x 64(e) tile.
// A tile staged in LDS as bf16 [n][c] so A-fragments are two ds_load_b128.
// ---------------------------------------------------------------------------
__global__ void __launch_bounds__(256)
proj_kernel(const float* __restrict__ in, const float* __restrict__ W,
            const float* __restrict__ bias, bf16_t* __restrict__ out,
            int transposed) {
    __shared__ __align__(16) bf16_t Abuf[64 * 40];   // [n][c], stride 40 elems (80B)

    const int tid  = threadIdx.x;
    const int lane = tid & 31;
    const int w    = tid >> 5;
    const int b    = blockIdx.z;
    const int n0   = blockIdx.x * 64;
    const int e0   = blockIdx.y * 64;
    const int nsub0 = (w & 3) * 16;
    const int es0   = e0 + (w >> 2) * 32;

    const float* inb = in + (size_t)b * CDIM * NTOK;

    v8f acc0 = {0,0,0,0,0,0,0,0};
    v8f acc1 = {0,0,0,0,0,0,0,0};

    const int m    = lane & 15;
    const int cb8  = (lane < 16) ? 0 : 8;    // A-layout K offset
    const int kb16 = (lane < 16) ? 0 : 16;   // unused; B fragments also use cb8

    for (int kk = 0; kk < CDIM; kk += 32) {
        // Stage + transpose + convert A tile: 32 c-rows x 64 n floats
        for (int s = tid; s < 512; s += 256) {
            int c  = s >> 4;
            int n4 = (s & 15) << 2;
            const float* src = inb + (size_t)(kk + c) * NTOK + n0 + n4;
            float4 v = *reinterpret_cast<const float4*>(src);
            if (kk + 32 < CDIM)
                __builtin_prefetch(src + (size_t)32 * NTOK, 0, 0);
            Abuf[(n4 + 0) * 40 + c] = (bf16_t)v.x;
            Abuf[(n4 + 1) * 40 + c] = (bf16_t)v.y;
            Abuf[(n4 + 2) * 40 + c] = (bf16_t)v.z;
            Abuf[(n4 + 3) * 40 + c] = (bf16_t)v.w;
        }
        __syncthreads();

        // A fragment: row n, K runs {cb8..cb8+7} and {cb8+16..cb8+23}
        const bf16_t* pa = &Abuf[(nsub0 + m) * 40 + cb8];
        v16bf a = ld_frag2(pa, pa + 16);

        // B fragments: W rows are K-contiguous in global
        const float* wp0 = W + (size_t)(es0 + m) * CDIM + kk + cb8;
        v16bf b0 = cvt16(wp0);
        v16bf b1 = cvt16(wp0 + 16 * CDIM);

        acc0 = wmma_bf16(a, b0, acc0);
        acc1 = wmma_bf16(a, b1, acc1);
        __syncthreads();
    }

    const int rofs = (lane < 16) ? 0 : 8;
    const float bb0 = bias[es0 + m];
    const float bb1 = bias[es0 + 16 + m];
#pragma unroll
    for (int r = 0; r < 8; ++r) {
        int n = n0 + nsub0 + r + rofs;
        {
            int e = es0 + m;
            int h = e >> 6, d = e & 63;
            size_t idx = transposed
                ? ((((size_t)b * NH + h) * HD + d) * NTOK + n)
                : ((((size_t)b * NH + h) * NTOK + n) * HD + d);
            out[idx] = (bf16_t)(acc0[r] + bb0);
        }
        {
            int e = es0 + 16 + m;
            int h = e >> 6, d = e & 63;
            size_t idx = transposed
                ? ((((size_t)b * NH + h) * HD + d) * NTOK + n)
                : ((((size_t)b * NH + h) * NTOK + n) * HD + d);
            out[idx] = (bf16_t)(acc1[r] + bb1);
        }
    }
}

// ---------------------------------------------------------------------------
// Flash attention, causal. One wave = 16 query rows; block = 8 waves = 128 rows,
// all on one (b,h). K/V tiles are DMA'd into double-buffered LDS by wave 0 via
// the Tensor Data Mover and consumed by all waves via ds_load_b128 fragments.
// q,k: (b,h,n,d) bf16 ; vT: (b,h,d,n) bf16 ; out: (b, e=h*64+d, n) f32.
// ---------------------------------------------------------------------------
__global__ void __launch_bounds__(256)
attn_kernel(const bf16_t* __restrict__ q, const bf16_t* __restrict__ k,
            const bf16_t* __restrict__ vT, float* __restrict__ out) {
    __shared__ __align__(16) bf16_t Kbuf[2][32 * 64];   // [key][d]   4KB x2
    __shared__ __align__(16) bf16_t Vbuf[2][64 * 32];   // [d][key]   4KB x2
    __shared__ __align__(16) bf16_t Pbuf[8][16 * 40];   // per-wave P tile

    const int tid  = threadIdx.x;
    const int lane = tid & 31;
    const int w    = tid >> 5;
    const int bh   = blockIdx.y;
    const int b    = bh >> 3;
    const int h    = bh & 7;
    const int q0   = blockIdx.x * 128 + w * 16;

    const bf16_t* qp = q  + (size_t)bh * NTOK * HD;
    const bf16_t* kp = k  + (size_t)bh * NTOK * HD;
    const bf16_t* vp = vT + (size_t)bh * HD * NTOK;
    bf16_t* P = Pbuf[w];

    const int m    = lane & 15;
    const int cb8  = (lane < 16) ? 0 : 8;
    const int kb16 = (lane < 16) ? 0 : 16;
    const int rofs = (lane < 16) ? 0 : 8;

    // Q fragments (A layout), resident: d in [0,32) and [32,64)
    v16bf qf[2];
#pragma unroll
    for (int f = 0; f < 2; ++f) {
        const bf16_t* p = qp + (size_t)(q0 + m) * HD + f * 32 + cb8;
        qf[f] = ld_frag2(p, p + 16);
    }

    v8f o[4];
#pragma unroll
    for (int t = 0; t < 4; ++t) o[t] = (v8f){0,0,0,0,0,0,0,0};
    float mrow[8], lrow[8];
#pragma unroll
    for (int r = 0; r < 8; ++r) { mrow[r] = -3.0e38f; lrow[r] = 0.f; }

    // Block-wide causal bound; waves past their own bound skip compute uniformly.
    const int nkb = ((blockIdx.x * 128 + 127) >> 5) + 1;
    int cur = 0;

    // Preload key-block 0 via TDM (wave 0 only; TDM ignores EXEC).
    if (w == 0) {
        tdm_load(tdm_g0(lds_off(&Kbuf[0][0]), (uint64_t)(size_t)kp),
                 tdm_g1(2048, 1, 2048, 0, 2048));                 // 32x64 linear
        tdm_load(tdm_g0(lds_off(&Vbuf[0][0]), (uint64_t)(size_t)vp),
                 tdm_g1(32, 64, 32, 64, NTOK));                   // 64 rows of 32
        __builtin_amdgcn_s_wait_tensorcnt(0);
    }
    __syncthreads();

    for (int kb = 0; kb < nkb; ++kb) {
        const int kn0 = kb << 5;

        // Kick off DMA for the next key block into the other buffer.
        if (w == 0 && kb + 1 < nkb) {
            const bf16_t* kn = kp + (size_t)(kn0 + 32) * HD;
            const bf16_t* vn = vp + (size_t)(kn0 + 32);
            tdm_load(tdm_g0(lds_off(&Kbuf[cur ^ 1][0]), (uint64_t)(size_t)kn),
                     tdm_g1(2048, 1, 2048, 0, 2048));
            tdm_load(tdm_g0(lds_off(&Vbuf[cur ^ 1][0]), (uint64_t)(size_t)vn),
                     tdm_g1(32, 64, 32, 64, NTOK));
        }

        if (kn0 <= q0 + 15) {   // wave-uniform causal guard
            const bf16_t* Kb = &Kbuf[cur][0];
            const bf16_t* Vb = &Vbuf[cur][0];

            // S = Q @ K^T
            v8f s0 = (v8f){0,0,0,0,0,0,0,0};
            v8f s1 = (v8f){0,0,0,0,0,0,0,0};
#pragma unroll
            for (int dc = 0; dc < 2; ++dc) {
                const bf16_t* pk0 = Kb + (size_t)m * HD + dc * 32 + cb8;
                const bf16_t* pk1 = Kb + (size_t)(16 + m) * HD + dc * 32 + cb8;
                v16bf bk0 = ld_frag2(pk0, pk0 + 16);
                v16bf bk1 = ld_frag2(pk1, pk1 + 16);
                s0 = wmma_bf16(qf[dc], bk0, s0);
                s1 = wmma_bf16(qf[dc], bk1, s1);
            }

            // Online softmax (row r in VGPR slot r across each 16-lane half)
            float alpha[8];
#pragma unroll
            for (int r = 0; r < 8; ++r) {
                const int qrow = q0 + r + rofs;
                float x0 = s0[r] * 0.125f;
                float x1 = s1[r] * 0.125f;
                if (kn0 + m > qrow)      x0 = -1.0e9f;
                if (kn0 + 16 + m > qrow) x1 = -1.0e9f;
                float mx = fmaxf(x0, x1);
                mx = fmaxf(mx, __shfl_xor(mx, 1, 32));
                mx = fmaxf(mx, __shfl_xor(mx, 2, 32));
                mx = fmaxf(mx, __shfl_xor(mx, 4, 32));
                mx = fmaxf(mx, __shfl_xor(mx, 8, 32));
                float mnew = fmaxf(mrow[r], mx);
                alpha[r] = exp2f((mrow[r] - mnew) * L2E);
                mrow[r] = mnew;
                float p0 = exp2f((x0 - mnew) * L2E);
                float p1 = exp2f((x1 - mnew) * L2E);
                float rs = p0 + p1;
                rs += __shfl_xor(rs, 1, 32);
                rs += __shfl_xor(rs, 2, 32);
                rs += __shfl_xor(rs, 4, 32);
                rs += __shfl_xor(rs, 8, 32);
                lrow[r] = lrow[r] * alpha[r] + rs;
                P[(r + rofs) * 40 + m]      = (bf16_t)p0;
                P[(r + rofs) * 40 + 16 + m] = (bf16_t)p1;
            }
#pragma unroll
            for (int r = 0; r < 8; ++r) {
                o[0][r] *= alpha[r]; o[1][r] *= alpha[r];
                o[2][r] *= alpha[r]; o[3][r] *= alpha[r];
            }

            // P as A fragment (wave-internal LDS round trip)
            const bf16_t* pp = P + m * 40 + cb8;
            v16bf pf = ld_frag2(pp, pp + 16);

            // O += P @ V
#pragma unroll
            for (int t = 0; t < 4; ++t) {
                const bf16_t* pv = Vb + (size_t)(t * 16 + m) * 32 + cb8;
                v16bf bv = ld_frag2(pv, pv + 16);
                o[t] = wmma_bf16(pf, bv, o[t]);
            }
        }

        __syncthreads();                       // all consumers done with buf[cur]
        if (w == 0 && kb + 1 < nkb)
            __builtin_amdgcn_s_wait_tensorcnt(0);   // next buffer landed
        __syncthreads();                       // release consumers
        cur ^= 1;
    }

    // Epilogue: normalize, write (b, h*64+d, n) f32
    float inv[8];
#pragma unroll
    for (int r = 0; r < 8; ++r) inv[r] = 1.0f / lrow[r];
#pragma unroll
    for (int t = 0; t < 4; ++t) {
        const int d = t * 16 + m;
        const size_t base = ((size_t)b * EDIM + h * 64 + d) * NTOK;
#pragma unroll
        for (int r = 0; r < 8; ++r) {
            out[base + q0 + r + rofs] = o[t][r] * inv[r];
        }
    }
}

extern "C" void kernel_launch(void* const* d_in, const int* in_sizes, int n_in,
                              void* d_out, int out_size, void* d_ws, size_t ws_size,
                              hipStream_t stream) {
    (void)in_sizes; (void)n_in; (void)out_size; (void)ws_size;
    const float* query = (const float*)d_in[0];
    const float* key   = (const float*)d_in[1];
    const float* Wq    = (const float*)d_in[2];
    const float* bq    = (const float*)d_in[3];
    const float* Wk    = (const float*)d_in[4];
    const float* bk    = (const float*)d_in[5];
    const float* Wv    = (const float*)d_in[6];
    const float* bv    = (const float*)d_in[7];
    float* out = (float*)d_out;

    const size_t per = (size_t)2 * NH * NTOK * HD;   // 4,194,304 bf16 each
    bf16_t* qws = (bf16_t*)d_ws;
    bf16_t* kws = qws + per;
    bf16_t* vws = kws + per;                          // 24 MB of d_ws

    dim3 pgrid(NTOK / 64, EDIM / 64, 2);
    proj_kernel<<<pgrid, 256, 0, stream>>>(query, Wq, bq, qws, 0);
    proj_kernel<<<pgrid, 256, 0, stream>>>(key,   Wk, bk, kws, 0);
    proj_kernel<<<pgrid, 256, 0, stream>>>(key,   Wv, bv, vws, 1);

    dim3 agrid(NTOK / 128, 2 * NH);
    attn_kernel<<<agrid, 256, 0, stream>>>(qws, kws, vws, out);
}
